// UAVGNN_41042707481180
// MI455X (gfx1250) — compile-verified, matrix operand order checked by reading, not run on hardware
//
#include <hip/hip_runtime.h>
#include <stdint.h>
#include <stddef.h>

// ---------------------------------------------------------------------------
// MI455X (gfx1250) GNN message-passing pipeline.
// Heavy matmuls on v_wmma_f32_16x16x32_bf16 (f32 accumulate).
// Edge kernel processes M=32 edges per wave: every B (weight) fragment loaded
// from L2 feeds two WMMAs -> 2x arithmetic intensity on the weight stream.
// ---------------------------------------------------------------------------

typedef __bf16 bf16_t;
typedef __attribute__((ext_vector_type(16))) __bf16 v16bf;
typedef __attribute__((ext_vector_type(8)))  __bf16 v8bf;
typedef __attribute__((ext_vector_type(4)))  __bf16 v4bf;
typedef __attribute__((ext_vector_type(8)))  float  v8f;

#define EWPB 2           // waves per block, edge kernel (M=32 tiles, 50KB LDS)
#define WPB 4            // waves per block, node/out kernels (M=16 tiles)
#define FSTRIDE 392      // LDS feat row stride in bf16 (784B = 49*16 -> conflict-free)
#define NSTRIDE 136      // LDS row stride for 128-col tiles (272B = 17*16)

__device__ __forceinline__ void lds_fence() {
  // same-wave LDS store->load handoff; also a compiler barrier
  asm volatile("s_wait_dscnt 0" ::: "memory");
}

__device__ __forceinline__ v8f wmma_bf16(const v16bf& a, const v16bf& b, const v8f& c) {
  return __builtin_amdgcn_wmma_f32_16x16x32_bf16(false, a, false, b, (short)0, c,
                                                 false, false);
}

__device__ __forceinline__ v4bf cvt4(float4 f) {
  v4bf t;
  t[0] = (bf16_t)f.x; t[1] = (bf16_t)f.y; t[2] = (bf16_t)f.z; t[3] = (bf16_t)f.w;
  return t;
}

__device__ __forceinline__ v16bf make_afrag(const bf16_t* Arow, int kt) {
  v8bf lo = *(const v8bf*)(Arow + (size_t)kt * 32);
  v8bf hv = *(const v8bf*)(Arow + (size_t)kt * 32 + 16);
  v16bf a;
  #pragma unroll
  for (int i = 0; i < 8; ++i) { a[i] = lo[i]; a[i + 8] = hv[i]; }
  return a;
}

// ---------------------------------------------------------------------------
// Weight pre-pack: f32 row-major [K x 128] -> bf16 B-fragment layout.
// Packed element t = ((kt*8 + nt)*32 + lane)*16 + p holds W[k][n] with
//   hi = lane>>4, nn = lane&15
//   kk = (p&7) + (hi?8:0) + (p>=8 ? 16:0), k = kt*32+kk, n = nt*16+nn
// (lanes 0-15 carry K 0-7 & 16-23, lanes 16-31 carry K 8-15 & 24-31; matches
//  the A-fragment pattern so a single convention is used end-to-end.)
// ---------------------------------------------------------------------------
__global__ void gnn_pack_kernel(const float* __restrict__ src,
                                bf16_t* __restrict__ dst, int ktiles) {
  int t = blockIdx.x * blockDim.x + threadIdx.x;
  int total = ktiles * 8 * 32 * 16;
  if (t >= total) return;
  int p    = t & 15;
  int lane = (t >> 4) & 31;
  int nt   = (t >> 9) & 7;
  int kt   = t >> 12;
  int hi   = lane >> 4;
  int nn   = lane & 15;
  int kk   = (p & 7) + (hi ? 8 : 0) + ((p >= 8) ? 16 : 0);
  int k    = kt * 32 + kk;
  int n    = nt * 16 + nn;
  dst[t] = (bf16_t)src[(size_t)k * 128 + n];
}

__global__ void gnn_zero_kernel(float* __restrict__ p, size_t n) {
  size_t t = (size_t)blockIdx.x * blockDim.x + threadIdx.x;
  if (t < n) p[t] = 0.0f;
}

__global__ void gnn_cvt_kernel(const float* __restrict__ src,
                               bf16_t* __restrict__ dst, size_t n) {
  size_t t = (size_t)blockIdx.x * blockDim.x + threadIdx.x;
  if (t < n) dst[t] = (bf16_t)src[t];
}

__global__ void gnn_deg_kernel(const int* __restrict__ ei,
                               float* __restrict__ cnt, int E) {
  int t = blockIdx.x * blockDim.x + threadIdx.x;
  if (t < E) atomicAdd(cnt + ei[E + t], 1.0f);   // dst = edge_index[1]
}

// M=32 WMMA pass: two stacked 16-row A sub-tiles share each B fragment.
__device__ __forceinline__ void wmma_pass2(const bf16_t* __restrict__ Arow0,
                                           const bf16_t* __restrict__ Arow1,
                                           const bf16_t* __restrict__ Wp,
                                           int lane, int ktiles,
                                           v8f acc0[8], v8f acc1[8]) {
  for (int kt = 0; kt < ktiles; ++kt) {
    v16bf a0 = make_afrag(Arow0, kt);
    v16bf a1 = make_afrag(Arow1, kt);
    const bf16_t* wp = Wp + ((size_t)(kt * 8) * 32 + lane) * 16;
    if (kt + 1 < ktiles)
      __builtin_prefetch(Wp + ((size_t)((kt + 1) * 8) * 32 + lane) * 16, 0, 1);
    #pragma unroll
    for (int nt = 0; nt < 8; ++nt) {
      v16bf b = *(const v16bf*)(wp + (size_t)nt * 32 * 16);
      acc0[nt] = wmma_bf16(a0, b, acc0[nt]);
      acc1[nt] = wmma_bf16(a1, b, acc1[nt]);
    }
  }
}

// M=16 WMMA pass (node/out kernels).
__device__ __forceinline__ void wmma_pass1(const bf16_t* __restrict__ Arow,
                                           const bf16_t* __restrict__ Wp,
                                           int lane, int ktiles, v8f acc[8]) {
  for (int kt = 0; kt < ktiles; ++kt) {
    v16bf a = make_afrag(Arow, kt);
    const bf16_t* wp = Wp + ((size_t)(kt * 8) * 32 + lane) * 16;
    if (kt + 1 < ktiles)
      __builtin_prefetch(Wp + ((size_t)((kt + 1) * 8) * 32 + lane) * 16, 0, 1);
    #pragma unroll
    for (int nt = 0; nt < 8; ++nt) {
      v16bf b = *(const v16bf*)(wp + (size_t)nt * 32 * 16);
      acc[nt] = wmma_bf16(a, b, acc[nt]);
    }
  }
}

// ---------------------------------------------------------------------------
// Edge MLP:  m = relu(feat @ W1 + b1) @ W2 + b2 ;  accum[dst] += m (atomic)
//   mode 0: feat = edge_attr                    (K = 128, ktiles = 4)
//   mode 1: feat = [h[dst], h[src], edge_attr]  (K = 384, ktiles = 12)
// One wave per 32-edge tile (two 16-row sub-tiles), 16 f32 accumulators.
// ---------------------------------------------------------------------------
__global__ __launch_bounds__(EWPB * 32)
void gnn_edge_mlp_kernel(const int* __restrict__ ei,
                         const float* __restrict__ eattr,
                         const bf16_t* __restrict__ hbf,
                         const bf16_t* __restrict__ W1p,
                         const float* __restrict__ b1,
                         const bf16_t* __restrict__ W2p,
                         const float* __restrict__ b2,
                         float* __restrict__ accum,
                         int E, int ktiles1, int mode) {
  __shared__ __attribute__((aligned(16))) bf16_t sfeat[EWPB][32][FSTRIDE];
  const int w = threadIdx.x >> 5;
  const int lane = threadIdx.x & 31;
  const int r0 = lane & 15;
  const int hi = lane >> 4;
  const int ntiles = (E + 31) >> 5;

  // tile-invariant bias values (col = nt*16 + r0), hoisted out of tile loop
  float bias1[8], bias2[8];
  #pragma unroll
  for (int nt = 0; nt < 8; ++nt) {
    bias1[nt] = b1[nt * 16 + r0];
    bias2[nt] = b2[nt * 16 + r0];
  }

  for (int tile = blockIdx.x * EWPB + w; tile < ntiles; tile += gridDim.x * EWPB) {
    const int e0 = tile << 5;
    bf16_t (&F)[32][FSTRIDE] = sfeat[w];

    // ---------------- gather 32-row feat tile into LDS (bf16) ----------------
    if (mode == 0) {
      // 64 chunks: (row, half) each 64 cols of edge_attr
      #pragma unroll
      for (int j = lane; j < 64; j += 32) {
        int row = j & 31, half = (j >> 5) & 1;
        int e = e0 + row;
        bf16_t* dp = &F[row][half * 64];
        if (e < E) {
          const float* sp = eattr + (size_t)e * 128 + half * 64;
          #pragma unroll
          for (int c = 0; c < 64; c += 4)
            *(v4bf*)(dp + c) = cvt4(*(const float4*)(sp + c));
        } else {
          v4bf z = {};
          #pragma unroll
          for (int c = 0; c < 64; c += 4) *(v4bf*)(dp + c) = z;
        }
      }
    } else {
      // 96 chunks of 128 cols: (row, source), source 0:h[dst] 1:h[src] 2:edge_attr
      #pragma unroll
      for (int j = lane; j < 96; j += 32) {
        int row = j & 31, s3 = j >> 5;
        int e = e0 + row;
        if (e < E) {
          if (s3 < 2) {
            int node = ei[(1 - s3) * E + e];   // s3==0 -> dst, s3==1 -> src
            const uint4* sp = (const uint4*)(hbf + (size_t)node * 128);
            uint4* dp = (uint4*)(&F[row][s3 * 128]);
            #pragma unroll
            for (int c = 0; c < 16; ++c) dp[c] = sp[c];
          } else {
            const float* sp = eattr + (size_t)e * 128;
            bf16_t* dp = &F[row][256];
            #pragma unroll
            for (int c = 0; c < 128; c += 4)
              *(v4bf*)(dp + c) = cvt4(*(const float4*)(sp + c));
          }
        } else {
          bf16_t* dp = &F[row][s3 * 128];
          v4bf z = {};
          #pragma unroll
          for (int c = 0; c < 128; c += 4) *(v4bf*)(dp + c) = z;
        }
      }
    }
    lds_fence();

    // ---------------- matmul 1: feat[32xK] @ W1[Kx128] ----------------
    const bf16_t* Arow0 = &F[r0][hi * 8];
    const bf16_t* Arow1 = &F[16 + r0][hi * 8];
    v8f zero8 = {0.f, 0.f, 0.f, 0.f, 0.f, 0.f, 0.f, 0.f};
    v8f acc0[8], acc1[8];
    #pragma unroll
    for (int nt = 0; nt < 8; ++nt) { acc0[nt] = zero8; acc1[nt] = zero8; }
    wmma_pass2(Arow0, Arow1, W1p, lane, ktiles1, acc0, acc1);

    // ------------- bias + relu -> intermediate 32x128 tile in LDS -------------
    // C/D layout: VGPR r -> row (r + 8*hi), col = lane&15 (+16*nt)
    #pragma unroll
    for (int nt = 0; nt < 8; ++nt) {
      int col = nt * 16 + r0;
      #pragma unroll
      for (int r = 0; r < 8; ++r) {
        float v0 = acc0[nt][r] + bias1[nt];
        float v1 = acc1[nt][r] + bias1[nt];
        v0 = v0 > 0.0f ? v0 : 0.0f;
        v1 = v1 > 0.0f ? v1 : 0.0f;
        F[r + hi * 8][col] = (bf16_t)v0;        // feat tile fully consumed above
        F[16 + r + hi * 8][col] = (bf16_t)v1;
      }
    }
    lds_fence();

    // ---------------- matmul 2: inter[32x128] @ W2[128x128] ----------------
    v8f acc2a[8], acc2b[8];
    #pragma unroll
    for (int nt = 0; nt < 8; ++nt) { acc2a[nt] = zero8; acc2b[nt] = zero8; }
    wmma_pass2(Arow0, Arow1, W2p, lane, 4, acc2a, acc2b);

    // ---------------- bias2 + scatter-add into accum[dst] ----------------
    #pragma unroll
    for (int r = 0; r < 8; ++r) {
      int row = r + hi * 8;
      {
        int e = e0 + row;
        if (e < E) {
          float* op = accum + (size_t)ei[E + e] * 128;
          #pragma unroll
          for (int nt = 0; nt < 8; ++nt)
            atomicAdd(op + nt * 16 + r0, acc2a[nt][r] + bias2[nt]);
        }
      }
      {
        int e = e0 + 16 + row;
        if (e < E) {
          float* op = accum + (size_t)ei[E + e] * 128;
          #pragma unroll
          for (int nt = 0; nt < 8; ++nt)
            atomicAdd(op + nt * 16 + r0, acc2b[nt][r] + bias2[nt]);
        }
      }
    }
  }
}

// ---------------------------------------------------------------------------
// Node update: h = relu((agg/deg) @ node_W + node_b); also bf16 mirror.
// ---------------------------------------------------------------------------
__global__ __launch_bounds__(WPB * 32)
void gnn_node_update_kernel(const float* __restrict__ agg,
                            const float* __restrict__ cnt,
                            const bf16_t* __restrict__ Wp,
                            const float* __restrict__ bias,
                            float* __restrict__ h,
                            bf16_t* __restrict__ hbf, int N) {
  __shared__ __attribute__((aligned(16))) bf16_t s[WPB][16][NSTRIDE];
  const int w = threadIdx.x >> 5;
  const int lane = threadIdx.x & 31;
  const int r0 = lane & 15;
  const int hi = lane >> 4;
  const int ntiles = (N + 15) >> 4;

  float bb[8];
  #pragma unroll
  for (int nt = 0; nt < 8; ++nt) bb[nt] = bias[nt * 16 + r0];

  for (int tile = blockIdx.x * WPB + w; tile < ntiles; tile += gridDim.x * WPB) {
    const int n0 = tile << 4;
    bf16_t (&F)[16][NSTRIDE] = s[w];

    { // stage (agg/deg) rows as bf16
      int row = r0, half = hi;
      int node = n0 + row;
      bf16_t* dp = &F[row][half * 64];
      if (node < N) {
        float inv = 1.0f / fmaxf(cnt[node], 1.0f);
        const float* sp = agg + (size_t)node * 128 + half * 64;
        #pragma unroll
        for (int c = 0; c < 64; c += 4) {
          float4 f = *(const float4*)(sp + c);
          f.x *= inv; f.y *= inv; f.z *= inv; f.w *= inv;
          *(v4bf*)(dp + c) = cvt4(f);
        }
      } else {
        v4bf z = {};
        #pragma unroll
        for (int c = 0; c < 64; c += 4) *(v4bf*)(dp + c) = z;
      }
    }
    lds_fence();

    const bf16_t* Arow = &F[r0][hi * 8];
    v8f zero8 = {0.f, 0.f, 0.f, 0.f, 0.f, 0.f, 0.f, 0.f};
    v8f acc[8];
    #pragma unroll
    for (int nt = 0; nt < 8; ++nt) acc[nt] = zero8;
    wmma_pass1(Arow, Wp, lane, 4, acc);

    #pragma unroll
    for (int nt = 0; nt < 8; ++nt) {
      int col = nt * 16 + r0;
      #pragma unroll
      for (int r = 0; r < 8; ++r) {
        int node = n0 + r + hi * 8;
        if (node < N) {
          float v = acc[nt][r] + bb[nt];
          v = v > 0.0f ? v : 0.0f;
          h[(size_t)node * 128 + col] = v;
          hbf[(size_t)node * 128 + col] = (bf16_t)v;
        }
      }
    }
  }
}

// ---------------------------------------------------------------------------
// Output head: out = relu(h @ out_W1 + b1) @ out_W2 + b2   (128 -> 128 -> 3)
// ---------------------------------------------------------------------------
__global__ __launch_bounds__(WPB * 32)
void gnn_out_kernel(const bf16_t* __restrict__ hbf,
                    const bf16_t* __restrict__ W1p,
                    const float* __restrict__ b1,
                    const float* __restrict__ W2,   // [128 x 3] f32
                    const float* __restrict__ b2,   // [3]
                    float* __restrict__ out, int N) {
  __shared__ __attribute__((aligned(16))) bf16_t s[WPB][16][NSTRIDE];
  const int w = threadIdx.x >> 5;
  const int lane = threadIdx.x & 31;
  const int r0 = lane & 15;
  const int hi = lane >> 4;
  const int ntiles = (N + 15) >> 4;

  float bb[8];
  #pragma unroll
  for (int nt = 0; nt < 8; ++nt) bb[nt] = b1[nt * 16 + r0];

  for (int tile = blockIdx.x * WPB + w; tile < ntiles; tile += gridDim.x * WPB) {
    const int n0 = tile << 4;
    bf16_t (&F)[16][NSTRIDE] = s[w];

    { // stage h rows (already bf16)
      int row = r0, half = hi;
      int node = n0 + row;
      uint4* dp = (uint4*)(&F[row][half * 64]);
      if (node < N) {
        const uint4* sp = (const uint4*)(hbf + (size_t)node * 128 + half * 64);
        #pragma unroll
        for (int c = 0; c < 8; ++c) dp[c] = sp[c];
      } else {
        uint4 z = {0u, 0u, 0u, 0u};
        #pragma unroll
        for (int c = 0; c < 8; ++c) dp[c] = z;
      }
    }
    lds_fence();

    const bf16_t* Arow = &F[r0][hi * 8];
    v8f zero8 = {0.f, 0.f, 0.f, 0.f, 0.f, 0.f, 0.f, 0.f};
    v8f acc[8];
    #pragma unroll
    for (int nt = 0; nt < 8; ++nt) acc[nt] = zero8;
    wmma_pass1(Arow, W1p, lane, 4, acc);

    // write relu(acc + b1) back to LDS (staging tile fully consumed)
    #pragma unroll
    for (int nt = 0; nt < 8; ++nt) {
      int col = nt * 16 + r0;
      #pragma unroll
      for (int r = 0; r < 8; ++r) {
        float v = acc[nt][r] + bb[nt];
        v = v > 0.0f ? v : 0.0f;
        F[r + hi * 8][col] = (bf16_t)v;
      }
    }
    lds_fence();

    // 128 -> 3 projection (tiny): lanes 0-15, one node row each
    if (lane < 16) {
      int node = n0 + lane;
      if (node < N) {
        float s0 = b2[0], s1 = b2[1], s2 = b2[2];
        #pragma unroll 8
        for (int k = 0; k < 128; ++k) {
          float x = (float)F[lane][k];
          s0 += x * W2[k * 3 + 0];
          s1 += x * W2[k * 3 + 1];
          s2 += x * W2[k * 3 + 2];
        }
        out[(size_t)node * 3 + 0] = s0;
        out[(size_t)node * 3 + 1] = s1;
        out[(size_t)node * 3 + 2] = s2;
      }
    }
  }
}

// ---------------------------------------------------------------------------
// Host-side orchestration (all launches on `stream`, graph-capture safe).
// ---------------------------------------------------------------------------
extern "C" void kernel_launch(void* const* d_in, const int* in_sizes, int n_in,
                              void* d_out, int out_size, void* d_ws, size_t ws_size,
                              hipStream_t stream) {
  const int*   ei     = (const int*)  d_in[1];
  const float* eattr  = (const float*)d_in[2];
  const float* e2nW1  = (const float*)d_in[3];
  const float* e2nb1  = (const float*)d_in[4];
  const float* e2nW2  = (const float*)d_in[5];
  const float* e2nb2  = (const float*)d_in[6];
  const float* edgeW1 = (const float*)d_in[7];
  const float* edgeb1 = (const float*)d_in[8];
  const float* edgeW2 = (const float*)d_in[9];
  const float* edgeb2 = (const float*)d_in[10];
  const float* nodeW  = (const float*)d_in[11];
  const float* nodeb  = (const float*)d_in[12];
  const float* outW1  = (const float*)d_in[13];
  const float* outb1  = (const float*)d_in[14];
  const float* outW2  = (const float*)d_in[15];
  const float* outb2  = (const float*)d_in[16];

  const int N = in_sizes[0] / 16;            // x: [N, 16] (only N is used)
  const int E = in_sizes[1] / 2;             // edge_index: [2, E]
  const int L = in_sizes[7] / (384 * 128);   // edge_W1: [L, 384, 128]

  // -------- workspace layout --------
  auto aup = [](size_t x) { return (x + 255) & ~(size_t)255; };
  char* ws = (char*)d_ws;
  size_t o = 0;
  const size_t PK4  = (size_t)4  * 8 * 32 * 16;   // packed 128x128 matrix (elems)
  const size_t PK12 = (size_t)12 * 8 * 32 * 16;   // packed 384x128 matrix (elems)

  bf16_t* hbf = (bf16_t*)(ws + o); o = aup(o + (size_t)N * 128 * sizeof(bf16_t));
  float*  h   = (float*) (ws + o); o = aup(o + (size_t)N * 128 * sizeof(float));
  float*  agg = (float*) (ws + o); o = aup(o + (size_t)N * 128 * sizeof(float));
  float*  cnt = (float*) (ws + o); o = aup(o + (size_t)N * sizeof(float));
  bf16_t* e2nW1p = (bf16_t*)(ws + o); o = aup(o + PK4 * sizeof(bf16_t));
  bf16_t* e2nW2p = (bf16_t*)(ws + o); o = aup(o + PK4 * sizeof(bf16_t));
  bf16_t* edgeW1p = (bf16_t*)(ws + o); o = aup(o + (size_t)L * PK12 * sizeof(bf16_t));
  bf16_t* edgeW2p = (bf16_t*)(ws + o); o = aup(o + (size_t)L * PK4 * sizeof(bf16_t));
  bf16_t* nodeWp  = (bf16_t*)(ws + o); o = aup(o + (size_t)L * PK4 * sizeof(bf16_t));
  bf16_t* outW1p  = (bf16_t*)(ws + o); o = aup(o + PK4 * sizeof(bf16_t));
  (void)ws_size; (void)n_in; (void)out_size;

  // -------- weight packing (f32 -> bf16 fragment layout) --------
  const int TB = 256;
  gnn_pack_kernel<<<(int)((PK4  + TB - 1) / TB), TB, 0, stream>>>(e2nW1, e2nW1p, 4);
  gnn_pack_kernel<<<(int)((PK4  + TB - 1) / TB), TB, 0, stream>>>(e2nW2, e2nW2p, 4);
  for (int l = 0; l < L; ++l) {
    gnn_pack_kernel<<<(int)((PK12 + TB - 1) / TB), TB, 0, stream>>>(
        edgeW1 + (size_t)l * 384 * 128, edgeW1p + (size_t)l * PK12, 12);
    gnn_pack_kernel<<<(int)((PK4 + TB - 1) / TB), TB, 0, stream>>>(
        edgeW2 + (size_t)l * 128 * 128, edgeW2p + (size_t)l * PK4, 4);
    gnn_pack_kernel<<<(int)((PK4 + TB - 1) / TB), TB, 0, stream>>>(
        nodeW + (size_t)l * 128 * 128, nodeWp + (size_t)l * PK4, 4);
  }
  gnn_pack_kernel<<<(int)((PK4 + TB - 1) / TB), TB, 0, stream>>>(outW1, outW1p, 4);

  // -------- degree + initial h = segment_sum(e2n_mlp(edge_attr), dst) --------
  const size_t NH = (size_t)N * 128;
  gnn_zero_kernel<<<(int)((NH + TB - 1) / TB), TB, 0, stream>>>(h, NH);
  gnn_zero_kernel<<<(int)((N + TB - 1) / TB), TB, 0, stream>>>(cnt, (size_t)N);
  gnn_deg_kernel<<<(E + TB - 1) / TB, TB, 0, stream>>>(ei, cnt, E);

  const int etiles = (E + 31) >> 5;
  const int eblocks = (etiles + EWPB - 1) / EWPB;
  gnn_edge_mlp_kernel<<<eblocks, EWPB * 32, 0, stream>>>(
      ei, eattr, (const bf16_t*)nullptr, e2nW1p, e2nb1, e2nW2p, e2nb2,
      h, E, /*ktiles1=*/4, /*mode=*/0);
  gnn_cvt_kernel<<<(int)((NH + TB - 1) / TB), TB, 0, stream>>>(h, hbf, NH);

  // -------- message-passing layers --------
  const int ntilesN = (N + 15) >> 4;
  const int nblocks = (ntilesN + WPB - 1) / WPB;
  for (int l = 0; l < L; ++l) {
    gnn_zero_kernel<<<(int)((NH + TB - 1) / TB), TB, 0, stream>>>(agg, NH);
    gnn_edge_mlp_kernel<<<eblocks, EWPB * 32, 0, stream>>>(
        ei, eattr, hbf,
        edgeW1p + (size_t)l * PK12, edgeb1 + (size_t)l * 128,
        edgeW2p + (size_t)l * PK4,  edgeb2 + (size_t)l * 128,
        agg, E, /*ktiles1=*/12, /*mode=*/1);
    gnn_node_update_kernel<<<nblocks, WPB * 32, 0, stream>>>(
        agg, cnt, nodeWp + (size_t)l * PK4, nodeb + (size_t)l * 128, h, hbf, N);
  }

  // -------- output head --------
  gnn_out_kernel<<<nblocks, WPB * 32, 0, stream>>>(
      hbf, outW1p, outb1, outW2, outb2, (float*)d_out, N);
}
